// LayoutEncoder_84559316124417
// MI455X (gfx1250) — matrix operand order repaired and checked by reading, not compiled, not run on hardware
//
#include <hip/hip_runtime.h>
#include <hip/hip_bf16.h>

typedef __attribute__((ext_vector_type(16))) _Float16 v16h;
typedef __attribute__((ext_vector_type(8)))  float    v8f;

// ---------------------------------------------------------------------------
// helpers
// ---------------------------------------------------------------------------
__device__ __forceinline__ float block_reduce_sum_256(float v, float* buf) {
    int t = threadIdx.x;
    buf[t] = v;
    __syncthreads();
    for (int s = 128; s > 0; s >>= 1) {
        if (t < s) buf[t] += buf[t + s];
        __syncthreads();
    }
    float r = buf[0];
    __syncthreads();
    return r;
}

// ---------------------------------------------------------------------------
// generic f32 -> f16 pack
// ---------------------------------------------------------------------------
__global__ __launch_bounds__(256) void pack_f16_kernel(const float* __restrict__ src,
                                                       _Float16* __restrict__ dst, int n) {
    int i = blockIdx.x * 256 + threadIdx.x;
    if (i < n) dst[i] = (_Float16)src[i];
}

// ---------------------------------------------------------------------------
// spatial MLP: bboxes(1024,4) -> LN/ReLU(256) -> LN(512), write f16 into
// A3[:, 0:512] (A3 row stride = 1024, concat buffer for the fusion GEMM)
// ---------------------------------------------------------------------------
__global__ __launch_bounds__(256) void spat_kernel(
    const float* __restrict__ bboxes,
    const float* __restrict__ sw1, const float* __restrict__ sb1,
    const float* __restrict__ sg1, const float* __restrict__ sbt1,
    const float* __restrict__ sw2, const float* __restrict__ sb2,
    const float* __restrict__ sg2, const float* __restrict__ sbt2,
    _Float16* __restrict__ A3) {
    __shared__ float sh_s[256];
    __shared__ float buf[256];
    const int row = blockIdx.x, t = threadIdx.x;
    const float b0 = bboxes[row * 4 + 0], b1 = bboxes[row * 4 + 1];
    const float b2 = bboxes[row * 4 + 2], b3 = bboxes[row * 4 + 3];

    float s = b0 * sw1[t] + b1 * sw1[256 + t] + b2 * sw1[512 + t] + b3 * sw1[768 + t] + sb1[t];
    float mean = block_reduce_sum_256(s, buf) * (1.0f / 256.0f);
    float d0 = s - mean;
    float var = block_reduce_sum_256(d0 * d0, buf) * (1.0f / 256.0f);
    float v = d0 * rsqrtf(var + 1e-5f) * sg1[t] + sbt1[t];
    v = fmaxf(v, 0.0f);
    sh_s[t] = v;
    __syncthreads();

    float a0 = sb2[t], a1 = sb2[t + 256];
    for (int k = 0; k < 256; k++) {
        float sv = sh_s[k];
        a0 += sv * sw2[k * 512 + t];
        a1 += sv * sw2[k * 512 + t + 256];
    }
    float m2 = block_reduce_sum_256(a0 + a1, buf) * (1.0f / 512.0f);
    float e0 = a0 - m2, e1 = a1 - m2;
    float v2 = block_reduce_sum_256(e0 * e0 + e1 * e1, buf) * (1.0f / 512.0f);
    float ri = rsqrtf(v2 + 1e-5f);
    float o0 = e0 * ri * sg2[t] + sbt2[t];
    float o1 = e1 * ri * sg2[t + 256] + sbt2[t + 256];
    A3[(size_t)row * 1024 + t]       = (_Float16)o0;
    A3[(size_t)row * 1024 + t + 256] = (_Float16)o1;
}

// ---------------------------------------------------------------------------
// ROI align one scale: one block per (roi, bin); 4 bilinear samples averaged.
// Writes f16 activation matrix A1 (16384 x 512).
// ---------------------------------------------------------------------------
__global__ __launch_bounds__(256) void roi_align_kernel(
    const float* __restrict__ feats,    // (4,28,28,512)
    const float* __restrict__ boxes,    // (1024,4)
    float scale, _Float16* __restrict__ Adst) {
    const int H = 28, W = 28, D = 512;
    const int pos = blockIdx.x & 15;
    const int roi = blockIdx.x >> 4;
    const int iy = pos >> 2, ix = pos & 3;
    const int bidx = roi >> 8;          // N = 256 rois per batch

    const float x1 = boxes[roi * 4 + 0] * scale - 0.5f;
    const float y1 = boxes[roi * 4 + 1] * scale - 0.5f;
    const float bw = (boxes[roi * 4 + 2] * scale - 0.5f - x1) * 0.25f;
    const float bh = (boxes[roi * 4 + 3] * scale - 0.5f - y1) * 0.25f;

    int   base[4][4];
    float wgt[4][4];
    int s = 0;
    for (int sy = 0; sy < 2; sy++) {
        for (int sx = 0; sx < 2; sx++, s++) {
            float y = y1 + ((float)iy + (sy + 0.5f) * 0.5f) * bh;
            float x = x1 + ((float)ix + (sx + 0.5f) * 0.5f) * bw;
            bool empty = (y < -1.0f) || (y > (float)H) || (x < -1.0f) || (x > (float)W);
            float yc = fminf(fmaxf(y, 0.0f), (float)(H - 1));
            float xc = fminf(fmaxf(x, 0.0f), (float)(W - 1));
            int yl = (int)floorf(yc); yl = yl < 0 ? 0 : (yl > H - 1 ? H - 1 : yl);
            int xl = (int)floorf(xc); xl = xl < 0 ? 0 : (xl > W - 1 ? W - 1 : xl);
            int yh = yl + 1 > H - 1 ? H - 1 : yl + 1;
            int xh = xl + 1 > W - 1 ? W - 1 : xl + 1;
            float ly = yc - (float)yl, lx = xc - (float)xl;
            float hy = 1.0f - ly, hx = 1.0f - lx;
            base[s][0] = ((bidx * H + yl) * W + xl) * D;
            base[s][1] = ((bidx * H + yl) * W + xh) * D;
            base[s][2] = ((bidx * H + yh) * W + xl) * D;
            base[s][3] = ((bidx * H + yh) * W + xh) * D;
            float q = empty ? 0.0f : 0.25f;    // mean over SRxSR samples
            wgt[s][0] = hy * hx * q;
            wgt[s][1] = hy * lx * q;
            wgt[s][2] = ly * hx * q;
            wgt[s][3] = ly * lx * q;
        }
    }
    for (int c = threadIdx.x; c < D; c += 256) {
        float v = 0.0f;
        for (int sp = 0; sp < 4; sp++)
            for (int t = 0; t < 4; t++)
                v += feats[base[sp][t] + c] * wgt[sp][t];
        Adst[((size_t)roi * 16 + pos) * 512 + c] = (_Float16)v;
    }
}

// ---------------------------------------------------------------------------
// Tiled WMMA GEMM: C(MxN,f32) = A(MxK,f16) * B(KxN,f16) + bias(N)
// 128x128 block tile, K-tile 64 (two 32-deep WMMA sub-steps per barrier).
// LDS holds tiles pre-shuffled into a per-lane WMMA fragment layout
// (k = 16*(lane/16) + e for A and B alike), so each fragment is ONE contiguous 32B chunk per lane
// -> two ds_load_b128 per v16h fragment, conflict-free.
// M,N multiples of 128; K multiple of 64 (true for all call sites).
// ---------------------------------------------------------------------------
#define TM 128
#define TN 128
#define TKW 64
__global__ __launch_bounds__(256) void gemm_f16_kernel(
    const _Float16* __restrict__ A, const _Float16* __restrict__ Bm,
    const float* __restrict__ bias, float* __restrict__ C,
    int M, int N, int K) {
    // [ksub(2)][tile(8)][lane(32)][8 dwords = 16 halfs]
    __shared__ alignas(32) unsigned int AsU[2 * 8 * 32 * 8];
    __shared__ alignas(32) unsigned int BsU[2 * 8 * 32 * 8];
    const int tid  = threadIdx.x;
    const int wave = tid >> 5;
    const int lane = tid & 31;
    const int bm = blockIdx.y * TM;
    const int bn = blockIdx.x * TN;

    v8f acc[8];
    for (int i = 0; i < 8; i++)
        for (int e = 0; e < 8; e++) acc[i][e] = 0.0f;

    const unsigned int* Aw = (const unsigned int*)A;   // global A as packed half-pairs

    for (int k0 = 0; k0 < K; k0 += TKW) {
        // ---- stage A tile (128 x 64 halfs) into WMMA A-fragment layout.
        // A 16x32 layout: lane = (m&15) + 16*(k>>4); half e = k&15.
        // k-pairs (2j,2j+1) share a lane and adjacent halves -> one b32 store.
        for (int i = tid; i < 4096; i += 256) {
            int m = i >> 5, j = i & 31;                // k = 2j in [0,64)
            unsigned int val = Aw[((size_t)(bm + m) * K + k0) / 2 + j];
            int ksub = j >> 4;                         // which 32-deep sub-tile
            int jj   = j & 15;                         // half-pair within sub-tile
            int lanew = (m & 15) + ((jj >> 3) << 4);
            int dword = jj & 7;
            AsU[(((ksub << 3) + (m >> 4)) * 32 + lanew) * 8 + dword] = val;
        }
        // ---- stage B tile (64 x 128 halfs) into WMMA B-fragment layout.
        // B 32x16 layout: lane = (n&15) + 16*(k>>4); half e = k&15.
        for (int i = tid; i < 4096; i += 256) {
            int n = i & 127, j = i >> 7;               // k = 2j in [0,64)
            union { _Float16 h[2]; unsigned int u; } cvt;
            cvt.h[0] = Bm[(size_t)(k0 + 2 * j) * N + bn + n];
            cvt.h[1] = Bm[(size_t)(k0 + 2 * j + 1) * N + bn + n];
            int ksub = j >> 4;
            int jj   = j & 15;                         // k' = 2*jj within sub-tile
            int lanew = (n & 15) + ((jj >> 3) << 4);
            int dword = jj & 7;
            BsU[(((ksub << 3) + (n >> 4)) * 32 + lanew) * 8 + dword] = cvt.u;
        }
        // ---- prefetch next K-tile while this one is consumed
        if (k0 + TKW < K) {
            int m = tid >> 1;
            __builtin_prefetch(A + (size_t)(bm + m) * K + k0 + TKW + (tid & 1) * 32, 0, 0);
            __builtin_prefetch(Bm + (size_t)(k0 + TKW + (tid >> 2)) * N + bn + (tid & 3) * 32, 0, 0);
        }
        __syncthreads();

        const v16h* AsF = (const v16h*)AsU;
        const v16h* BsF = (const v16h*)BsU;
        for (int ksub = 0; ksub < 2; ksub++) {
            v16h afrag = AsF[((ksub << 3) + wave) * 32 + lane];   // 2x ds_load_b128
            for (int nt = 0; nt < 8; nt++) {
                v16h bfrag = BsF[((ksub << 3) + nt) * 32 + lane]; // 2x ds_load_b128
                acc[nt] = __builtin_amdgcn_wmma_f32_16x16x32_f16(
                    false, afrag, false, bfrag, (short)0, acc[nt], false, false);
            }
        }
        __syncthreads();
    }

    // C/D layout: VGPR j -> (M = j + 8*(lane>=16), N = lane%16) within 16x16 tile
    const int halfSel = lane >> 4;
    const int lmod = lane & 15;
    for (int nt = 0; nt < 8; nt++) {
        const int nn = bn + nt * 16 + lmod;
        const float bv = bias ? bias[nn] : 0.0f;
        for (int j = 0; j < 8; j++) {
            const int mm = bm + wave * 16 + halfSel * 8 + j;
            C[(size_t)mm * N + nn] = acc[nt][j] + bv;
        }
    }
}

// ---------------------------------------------------------------------------
// Per-ROI LayerNorm over (4,4,512)=8192 elems, affine (pg/pb), ReLU,
// running max across scales.
// ---------------------------------------------------------------------------
__global__ __launch_bounds__(256) void ln_roi_max_kernel(
    const float* __restrict__ conv,     // (16384, 512); row = roi*16 + hw
    const float* __restrict__ pg,       // (512, 16) for this scale: g[h][w][d]=pg[d*16+hw]
    const float* __restrict__ pb,
    float* __restrict__ roimax, int first) {
    __shared__ float buf[256];
    const int roi = blockIdx.x;
    const float* src = conv + (size_t)roi * 8192;
    float s = 0.0f, s2 = 0.0f;
    for (int e = threadIdx.x; e < 8192; e += 256) {
        float v = src[e];
        s += v; s2 += v * v;
    }
    float mean = block_reduce_sum_256(s, buf) * (1.0f / 8192.0f);
    float msq  = block_reduce_sum_256(s2, buf) * (1.0f / 8192.0f);
    float ri = rsqrtf(msq - mean * mean + 1e-5f);
    for (int e = threadIdx.x; e < 8192; e += 256) {
        int hw = e >> 9, d = e & 511;
        float v = (src[e] - mean) * ri * pg[d * 16 + hw] + pb[d * 16 + hw];
        v = fmaxf(v, 0.0f);
        float* dst = roimax + (size_t)roi * 8192 + e;
        *dst = first ? v : fmaxf(*dst, v);
    }
}

// ---------------------------------------------------------------------------
// Pack A2 for the ROI FC: transpose (r,hw,d) -> (r, d*16+hw) as f16
// ---------------------------------------------------------------------------
__global__ __launch_bounds__(256) void packA2_kernel(const float* __restrict__ roimax,
                                                     _Float16* __restrict__ A2) {
    const int r = blockIdx.x;
    for (int e = threadIdx.x; e < 8192; e += 256) {
        int hw = e >> 9, d = e & 511;
        A2[(size_t)r * 8192 + d * 16 + hw] = (_Float16)roimax[(size_t)r * 8192 + e];
    }
}

// ---------------------------------------------------------------------------
// Per-row LayerNorm over 512, optional ReLU; writes f32 and/or f16 outputs.
// ---------------------------------------------------------------------------
__global__ __launch_bounds__(256) void ln512_kernel(
    const float* __restrict__ src, const float* __restrict__ gamma,
    const float* __restrict__ beta, int do_relu,
    float* __restrict__ out_f32, _Float16* __restrict__ out_f16,
    int f16_ld, int f16_off) {
    __shared__ float buf[256];
    const int row = blockIdx.x, t = threadIdx.x;
    const float a0 = src[(size_t)row * 512 + t];
    const float a1 = src[(size_t)row * 512 + t + 256];
    float mean = block_reduce_sum_256(a0 + a1, buf) * (1.0f / 512.0f);
    float e0 = a0 - mean, e1 = a1 - mean;
    float var = block_reduce_sum_256(e0 * e0 + e1 * e1, buf) * (1.0f / 512.0f);
    float ri = rsqrtf(var + 1e-5f);
    float o0 = e0 * ri * gamma[t] + beta[t];
    float o1 = e1 * ri * gamma[t + 256] + beta[t + 256];
    if (do_relu) { o0 = fmaxf(o0, 0.0f); o1 = fmaxf(o1, 0.0f); }
    if (out_f32) {
        out_f32[(size_t)row * 512 + t]       = o0;
        out_f32[(size_t)row * 512 + t + 256] = o1;
    }
    if (out_f16) {
        out_f16[(size_t)row * f16_ld + f16_off + t]       = (_Float16)o0;
        out_f16[(size_t)row * f16_ld + f16_off + t + 256] = (_Float16)o1;
    }
}

// ---------------------------------------------------------------------------
// launcher
// ---------------------------------------------------------------------------
extern "C" void kernel_launch(void* const* d_in, const int* in_sizes, int n_in,
                              void* d_out, int out_size, void* d_ws, size_t ws_size,
                              hipStream_t stream) {
    const float* bboxes = (const float*)d_in[0];
    const float* vis    = (const float*)d_in[1];
    const float* sw1  = (const float*)d_in[2];
    const float* sb1  = (const float*)d_in[3];
    const float* sg1  = (const float*)d_in[4];
    const float* sbt1 = (const float*)d_in[5];
    const float* sw2  = (const float*)d_in[6];
    const float* sb2  = (const float*)d_in[7];
    const float* sg2  = (const float*)d_in[8];
    const float* sbt2 = (const float*)d_in[9];
    const float* conv_w = (const float*)d_in[10];
    const float* conv_b = (const float*)d_in[11];
    const float* pg  = (const float*)d_in[12];
    const float* pb  = (const float*)d_in[13];
    const float* rw  = (const float*)d_in[14];
    const float* rb  = (const float*)d_in[15];
    const float* rg  = (const float*)d_in[16];
    const float* rbt = (const float*)d_in[17];
    const float* fw  = (const float*)d_in[18];
    const float* fb  = (const float*)d_in[19];
    const float* fg  = (const float*)d_in[20];
    const float* fbt = (const float*)d_in[21];
    (void)in_sizes; (void)n_in; (void)out_size; (void)ws_size;

    char* ws = (char*)d_ws;
    size_t off = 0;
    auto alloc = [&](size_t bytes) -> void* {
        void* p = ws + off;
        off += (bytes + 255) & ~(size_t)255;
        return p;
    };
    _Float16* A1   = (_Float16*)alloc((size_t)16384 * 512 * 2);
    float*    conv = (float*)   alloc((size_t)16384 * 512 * 4);
    float*    rmax = (float*)   alloc((size_t)16384 * 512 * 4);
    _Float16* A2   = (_Float16*)alloc((size_t)1024 * 8192 * 2);
    _Float16* A3   = (_Float16*)alloc((size_t)1024 * 1024 * 2);
    _Float16* rw16 = (_Float16*)alloc((size_t)8192 * 512 * 2);
    _Float16* cw16 = (_Float16*)alloc((size_t)3 * 512 * 512 * 2);
    _Float16* fw16 = (_Float16*)alloc((size_t)1024 * 512 * 2);
    float*    g2   = (float*)   alloc((size_t)1024 * 512 * 4);
    float*    g3   = (float*)   alloc((size_t)1024 * 512 * 4);

    // pack weights to f16 (deterministic per call; graph-safe)
    pack_f16_kernel<<<(8192 * 512 + 255) / 256, 256, 0, stream>>>(rw, rw16, 8192 * 512);
    pack_f16_kernel<<<(3 * 512 * 512 + 255) / 256, 256, 0, stream>>>(conv_w, cw16, 3 * 512 * 512);
    pack_f16_kernel<<<(1024 * 512 + 255) / 256, 256, 0, stream>>>(fw, fw16, 1024 * 512);

    // spatial branch -> f16 into A3[:, 0:512]
    spat_kernel<<<1024, 256, 0, stream>>>(bboxes, sw1, sb1, sg1, sbt1,
                                          sw2, sb2, sg2, sbt2, A3);

    // ROI branch: 3 scales, running max
    const float scales[3] = {0.125f, 0.0625f, 0.03125f};
    for (int i = 0; i < 3; i++) {
        roi_align_kernel<<<16384, 256, 0, stream>>>(vis, bboxes, scales[i], A1);
        gemm_f16_kernel<<<dim3(512 / TN, 16384 / TM), 256, 0, stream>>>(
            A1, cw16 + (size_t)i * 512 * 512, conv_b + i * 512, conv, 16384, 512, 512);
        ln_roi_max_kernel<<<1024, 256, 0, stream>>>(
            conv, pg + (size_t)i * 512 * 16, pb + (size_t)i * 512 * 16, rmax, i == 0 ? 1 : 0);
    }

    // ROI FC: (1024 x 8192) @ (8192 x 512) -> LN -> f16 into A3[:, 512:1024]
    packA2_kernel<<<1024, 256, 0, stream>>>(rmax, A2);
    gemm_f16_kernel<<<dim3(512 / TN, 1024 / TM), 256, 0, stream>>>(
        A2, rw16, rb, g2, 1024, 512, 8192);
    ln512_kernel<<<1024, 256, 0, stream>>>(g2, rg, rbt, 0, nullptr, A3, 1024, 512);

    // fusion FC: (1024 x 1024) @ (1024 x 512) -> LN -> ReLU -> d_out (f32)
    gemm_f16_kernel<<<dim3(512 / TN, 1024 / TM), 256, 0, stream>>>(
        A3, fw16, fb, g3, 1024, 512, 1024);
    ln512_kernel<<<1024, 256, 0, stream>>>(g3, fg, fbt, 1, (float*)d_out, nullptr, 0, 0);
}
